// StateActionValueAutoregressiveV2_64390149701734
// MI455X (gfx1250) — compile-verified, matrix operand order checked by reading, not run on hardware
//
#include <hip/hip_runtime.h>
#include <hip/hip_bf16.h>

typedef __attribute__((ext_vector_type(16))) __bf16 v16bf;
typedef __attribute__((ext_vector_type(8)))  __bf16 v8bf;
typedef __attribute__((ext_vector_type(8)))  float  v8f;

// Problem constants (match reference)
constexpr int B_    = 128;
constexpr int OBS_  = 512;
constexpr int ACT_  = 7;
constexpr int C_    = 100;
constexpr int CTX_  = 128;
constexpr int H_    = 512;
constexpr int IN_   = OBS_ + 1 + CTX_;   // 641
constexpr int KIN_  = OBS_ + CTX_;       // 640 (action column handled rank-1)
constexpr int NCAND = C_ + 1;            // 100 candidates + 1 actual-action row
constexpr int R_    = B_ * NCAND;        // 12928 rows (divisible by 64)
constexpr int QW_   = (ACT_ - 1) * C_;   // 600 written q columns
constexpr int QTOT_ = ACT_ * C_;         // 700 total q columns (100 implicit zeros)

// LDS tiling for the WMMA GEMM: 64x32 A-tile + 128x32 B-tile, rows padded to 40
// bf16 (80 B = 20 dwords) so a 32-lane ds_load_b128 fragment read tiles all 64
// banks conflict-free. Double buffered: 2 * (64+128)*40 * 2B = 30 KB.
constexpr int LDT_   = 40;               // padded row stride (bf16 elements)
constexpr int BOFF_  = 64 * LDT_;        // B-tile offset within a stage (elements)
constexpr int STAGE_ = (64 + 128) * LDT_; // one stage (elements)

#define WAIT_ASYNC_3() asm volatile("s_wait_asynccnt 0x3" ::: "memory")
#define WAIT_ASYNC_0() asm volatile("s_wait_asynccnt 0x0" ::: "memory")

__device__ __forceinline__ __bf16 f2bf(float f) {
  unsigned u = __builtin_bit_cast(unsigned, f);
  u += 0x7FFFu + ((u >> 16) & 1u);           // round-to-nearest-even
  unsigned short h = (unsigned short)(u >> 16);
  return __builtin_bit_cast(__bf16, h);
}

// One async 16-byte global->LDS transfer per lane (ASYNCcnt-tracked, GV mode).
__device__ __forceinline__ void async_b128(unsigned lds_off, const void* g) {
  asm volatile("global_load_async_to_lds_b128 %0, %1, off"
               :: "v"(lds_off), "v"((unsigned long long)(size_t)g)
               : "memory");
}

// Load a 16x32 bf16 WMMA fragment (A or B^T), row-major source, leading dim ld.
// ISA layout: lanes 0-15 hold row (lane), K chunks {0..7, 16..23};
//             lanes 16-31 hold row (lane-16), K chunks {8..15, 24..31}.
__device__ __forceinline__ v16bf load_frag(const __bf16* __restrict__ p, int ld,
                                           int row0, int k0, int lane) {
  const __bf16* rp = p + (size_t)(row0 + (lane & 15)) * ld + k0 + ((lane >> 4) << 3);
  v8bf lo = *(const v8bf*)rp;
  v8bf hi = *(const v8bf*)(rp + 16);
  v16bf r;
#pragma unroll
  for (int t = 0; t < 8; ++t) { r[t] = lo[t]; r[t + 8] = hi[t]; }
  return r;
}

// Issue one double-buffer stage: 256 threads cooperatively fetch the 64x32 A
// tile (256 b128 transfers) and 128x32 B tile (512 transfers) = 3 async ops
// per thread, written into padded LDS rows.
__device__ __forceinline__ void issue_stage(const __bf16* __restrict__ A,
                                            const __bf16* __restrict__ Bt,
                                            int K, int blkRow, int blkCol,
                                            unsigned ldsbase, int s, int k0,
                                            int tid) {
  const int arow = tid >> 2, aseg = tid & 3;
  async_b128(ldsbase + (unsigned)((s * STAGE_ + arow * LDT_ + aseg * 8) * 2),
             A + (size_t)(blkRow + arow) * K + k0 + aseg * 8);
#pragma unroll
  for (int h = 0; h < 2; ++h) {
    const int id = tid + h * 256;
    const int brow = id >> 2, bseg = id & 3;
    async_b128(ldsbase + (unsigned)((s * STAGE_ + BOFF_ + brow * LDT_ + bseg * 8) * 2),
               Bt + (size_t)(blkCol + brow) * K + k0 + bseg * 8);
  }
}

// C = A[M,K](bf16,row-major) x B^T[N,K](bf16,N-major) + bias[N]; optional relu;
// output f32 (Cf) or bf16 (Cb). Block = 256 threads = 8 waves; wave -> 16x64
// strip; workgroup -> 64x128 tile, async-LDS double-buffered over K in steps
// of 32. All problem dims divide the tile sizes (no bounds checks).
template <int RELU, int OUTBF>
__global__ __launch_bounds__(256) void gemm_bf16_wmma(
    const __bf16* __restrict__ A, const __bf16* __restrict__ Bt,
    const float* __restrict__ bias, float* __restrict__ Cf,
    __bf16* __restrict__ Cb, int M, int N, int K) {
  __shared__ alignas(16) __bf16 lds[2 * STAGE_];
  const int tid    = threadIdx.x;
  const int lane   = tid & 31;
  const int wave   = tid >> 5;
  const int blkRow = blockIdx.y * 64;
  const int blkCol = blockIdx.x * 128;
  const int rowLoc = (wave & 3) * 16;
  const int colLoc = (wave >> 2) * 64;
  const unsigned ldsbase = (unsigned)(size_t)&lds[0];
  (void)M;
  v8f acc[4] = {};
  const int nk = K >> 5;
  issue_stage(A, Bt, K, blkRow, blkCol, ldsbase, 0, 0, tid);
  for (int ks = 0; ks < nk; ++ks) {
    const int s = ks & 1;
    if (ks + 1 < nk) {
      issue_stage(A, Bt, K, blkRow, blkCol, ldsbase, s ^ 1, (ks + 1) << 5, tid);
      WAIT_ASYNC_3();   // only the just-issued stage may remain outstanding
    } else {
      WAIT_ASYNC_0();
    }
    __syncthreads();    // all waves' async writes for stage s visible
    const __bf16* la = lds + s * STAGE_ + rowLoc * LDT_;
    const __bf16* lb = lds + s * STAGE_ + BOFF_ + colLoc * LDT_;
    v16bf a = load_frag(la, LDT_, 0, 0, lane);
#pragma unroll
    for (int t = 0; t < 4; ++t) {
      v16bf b = load_frag(lb, LDT_, t * 16, 0, lane);
      acc[t] = __builtin_amdgcn_wmma_f32_16x16x32_bf16(
          false, a, false, b, (short)0, acc[t], false, false);
    }
    __syncthreads();    // everyone done with stage s before it is overwritten
  }
  // C/D layout: lanes 0-15 -> N=lane, M=+e ; lanes 16-31 -> N=lane-16, M=+8+e
  const int nc   = blkCol + colLoc + (lane & 15);
  const int rsel = (lane >> 4) << 3;
#pragma unroll
  for (int t = 0; t < 4; ++t) {
    const int n  = nc + t * 16;
    const float bv = bias[n];
#pragma unroll
    for (int e = 0; e < 8; ++e) {
      const int m = blkRow + rowLoc + rsel + e;
      float v = acc[t][e] + bv;
      if (RELU) v = fmaxf(v, 0.f);
      if (OUTBF) Cb[(size_t)m * N + n] = f2bf(v);
      else       Cf[(size_t)m * N + n] = v;
    }
  }
}

// ---- prep kernels: transpose+convert weights to bf16 (N-major), misc init ----
__global__ void prep_w1t(const float* __restrict__ W1, __bf16* __restrict__ W1t) {
  int idx = blockIdx.x * blockDim.x + threadIdx.x;
  if (idx >= ACT_ * H_ * KIN_) return;
  int i = idx / (H_ * KIN_), rem = idx % (H_ * KIN_);
  int n = rem / KIN_, k = rem % KIN_;
  int kk = (k < OBS_) ? k : k + 1;  // skip the action row (handled rank-1)
  W1t[idx] = f2bf(W1[((size_t)i * IN_ + kk) * H_ + n]);
}

__global__ void prep_w2t(const float* __restrict__ W2, __bf16* __restrict__ W2t) {
  int idx = blockIdx.x * blockDim.x + threadIdx.x;
  if (idx >= ACT_ * H_ * H_) return;
  int i = idx / (H_ * H_), rem = idx % (H_ * H_);
  int n = rem / H_, k = rem % H_;
  W2t[idx] = f2bf(W2[((size_t)i * H_ + k) * H_ + n]);
}

__global__ void prep_w3t(const float* __restrict__ W3, __bf16* __restrict__ W3t) {
  int idx = blockIdx.x * blockDim.x + threadIdx.x;
  if (idx >= ACT_ * CTX_ * H_) return;
  int i = idx / (CTX_ * H_), rem = idx % (CTX_ * H_);
  int n = rem / H_, k = rem % H_;
  W3t[idx] = f2bf(W3[((size_t)i * H_ + k) * CTX_ + n]);
}

__global__ void prep_misc(const float* __restrict__ obs, const float* __restrict__ acts,
                          const float* __restrict__ W1, __bf16* __restrict__ Abuf,
                          float* __restrict__ ctx0, float* __restrict__ actsd,
                          float* __restrict__ wact) {
  int idx = blockIdx.x * blockDim.x + threadIdx.x;  // 65536 threads
  if (idx < B_ * OBS_) {
    int b = idx / OBS_, k = idx % OBS_;
    Abuf[(size_t)b * KIN_ + k] = f2bf(obs[idx]);
  }
  if (idx < B_ * CTX_) {
    int b = idx / CTX_, c = idx % CTX_;
    Abuf[(size_t)b * KIN_ + OBS_ + c] = f2bf(0.f);  // context starts at zero
    ctx0[idx] = 0.f;
  }
  if (idx < B_ * ACT_) {
    float a = acts[idx];
    actsd[idx] = fminf(fmaxf(floorf((a + 1.f) * 0.5f * (float)C_), 0.f), (float)(C_ - 1));
  }
  if (idx < ACT_ * H_) {
    int i = idx / H_, n = idx % H_;
    wact[idx] = W1[((size_t)i * IN_ + OBS_) * H_ + n];  // action-scalar row of W1
  }
}

// h1[r,h] = relu(base1[b,h] + jval * wact[i,h]) -> bf16 ; jval = j (<C) or disc action
__global__ __launch_bounds__(256) void expand_h1(
    const float* __restrict__ base1, const float* __restrict__ wact,
    const float* __restrict__ actsd, __bf16* __restrict__ h1, int ai) {
  int idx = blockIdx.x * blockDim.x + threadIdx.x;  // R_ * (H_/8)
  int r  = idx >> 6;           // H_/8 == 64 vec8 groups per row
  int hb = (idx & 63) << 3;
  int b = r / NCAND, j = r % NCAND;
  float jv = (j < C_) ? (float)j : actsd[b * ACT_ + ai];
  const float* bp = base1 + (size_t)b * H_ + hb;
  const float* wp = wact + (size_t)ai * H_ + hb;
  v8bf o;
#pragma unroll
  for (int e = 0; e < 8; ++e) o[e] = f2bf(fmaxf(bp[e] + jv * wp[e], 0.f));
  *(v8bf*)(h1 + (size_t)r * H_ + hb) = o;
}

// Per-row epilogue of GEMM3: add context; j<C -> q = (h3+ctx)·Wp + bp via wave reduce;
// j==C -> context update (f32 ping-pong + bf16 into Abuf ctx columns for next layer1).
__global__ __launch_bounds__(256) void epilogue(
    const float* __restrict__ h3, const float* __restrict__ ctx_old,
    float* __restrict__ ctx_new, __bf16* __restrict__ Abuf,
    const float* __restrict__ Wp, const float* __restrict__ bp,
    float* __restrict__ q, int ai) {
  int wave = threadIdx.x >> 5, lane = threadIdx.x & 31;
  int r = blockIdx.x * 8 + wave;
  int b = r / NCAND, j = r % NCAND;
  const float* hr = h3 + (size_t)r * CTX_;
  const float* cr = ctx_old + (size_t)b * CTX_;
  if (j < C_) {
    float s = 0.f;
#pragma unroll
    for (int t = 0; t < 4; ++t) {
      int c = lane + 32 * t;
      s += (hr[c] + cr[c]) * Wp[c];
    }
#pragma unroll
    for (int off = 16; off > 0; off >>= 1) s += __shfl_xor(s, off, 32);
    if (lane == 0) q[(size_t)b * QW_ + (size_t)(ai - 1) * C_ + j] = s + bp[0];
  } else {
#pragma unroll
    for (int t = 0; t < 4; ++t) {
      int c = lane + 32 * t;
      float v = hr[c] + cr[c];
      ctx_new[(size_t)b * CTX_ + c] = v;
      Abuf[(size_t)b * KIN_ + OBS_ + c] = f2bf(v);
    }
  }
}

// logsumexp over 700 columns: 600 from q + 100 implicit zeros (reference never writes them)
__global__ void lse_kernel(const float* __restrict__ q, float* __restrict__ out) {
  int b = threadIdx.x;
  if (b >= B_) return;
  const float* qr = q + (size_t)b * QW_;
  float m = 0.f;  // implicit zero columns participate in max
  for (int c = 0; c < QW_; ++c) m = fmaxf(m, qr[c]);
  float s = (float)(QTOT_ - QW_) * expf(-m);
  for (int c = 0; c < QW_; ++c) s += expf(qr[c] - m);
  out[b] = m + logf(s);
}

extern "C" void kernel_launch(void* const* d_in, const int* in_sizes, int n_in,
                              void* d_out, int out_size, void* d_ws, size_t ws_size,
                              hipStream_t stream) {
  (void)in_sizes; (void)n_in; (void)out_size; (void)ws_size;
  const float* obs  = (const float*)d_in[0];
  const float* acts = (const float*)d_in[1];
  const float* W1   = (const float*)d_in[2];
  const float* b1   = (const float*)d_in[3];
  const float* W2   = (const float*)d_in[4];
  const float* b2   = (const float*)d_in[5];
  const float* W3   = (const float*)d_in[6];
  const float* b3   = (const float*)d_in[7];
  const float* Wp   = (const float*)d_in[8];
  const float* bp   = (const float*)d_in[9];
  float* out = (float*)d_out;

  // Workspace carve-out (~43 MB), 256B-aligned blocks
  char* ws = (char*)d_ws;
  size_t off = 0;
  auto alloc = [&](size_t bytes) {
    char* p = ws + off;
    off += (bytes + 255) & ~(size_t)255;
    return (void*)p;
  };
  __bf16* W1t   = (__bf16*)alloc((size_t)ACT_ * H_ * KIN_ * 2);
  __bf16* W2t   = (__bf16*)alloc((size_t)ACT_ * H_ * H_ * 2);
  __bf16* W3t   = (__bf16*)alloc((size_t)ACT_ * CTX_ * H_ * 2);
  float*  wact  = (float*) alloc((size_t)ACT_ * H_ * 4);
  __bf16* Abuf  = (__bf16*)alloc((size_t)B_ * KIN_ * 2);
  float*  base1 = (float*) alloc((size_t)B_ * H_ * 4);
  __bf16* h1    = (__bf16*)alloc((size_t)R_ * H_ * 2);
  __bf16* h2    = (__bf16*)alloc((size_t)R_ * H_ * 2);
  float*  h3    = (float*) alloc((size_t)R_ * CTX_ * 4);
  float*  ctxws = (float*) alloc((size_t)2 * B_ * CTX_ * 4);
  float*  qbuf  = (float*) alloc((size_t)B_ * QW_ * 4);
  float*  actsd = (float*) alloc((size_t)B_ * ACT_ * 4);
  float* ctxbuf[2] = { ctxws, ctxws + (size_t)B_ * CTX_ };

  // ---- prep ----
  {
    int n1 = ACT_ * H_ * KIN_;
    prep_w1t<<<(n1 + 255) / 256, 256, 0, stream>>>(W1, W1t);
    int n2 = ACT_ * H_ * H_;
    prep_w2t<<<(n2 + 255) / 256, 256, 0, stream>>>(W2, W2t);
    int n3 = ACT_ * CTX_ * H_;
    prep_w3t<<<(n3 + 255) / 256, 256, 0, stream>>>(W3, W3t);
    prep_misc<<<(B_ * OBS_) / 256, 256, 0, stream>>>(obs, acts, W1, Abuf,
                                                     ctxbuf[0], actsd, wact);
  }

  // ---- 6 autoregressive steps ----
  for (int i = 1; i < ACT_; ++i) {
    const int prev = (i - 1) & 1, cur = i & 1;
    // layer1: base1 = [obs|ctx] @ W1' + b1   ([128x640]x[640x512], f32 out)
    gemm_bf16_wmma<0, 0><<<dim3(H_ / 128, B_ / 64), 256, 0, stream>>>(
        Abuf, W1t + (size_t)i * H_ * KIN_, b1 + (size_t)i * H_,
        base1, nullptr, B_, H_, KIN_);
    // rank-1 action expansion + relu -> h1 bf16 [12928 x 512]
    expand_h1<<<(R_ * (H_ / 8)) / 256, 256, 0, stream>>>(base1, wact, actsd, h1, i);
    // layer2: h2 = relu(h1 @ W2 + b2)  ([12928x512]x[512x512], bf16 out)
    gemm_bf16_wmma<1, 1><<<dim3(H_ / 128, R_ / 64), 256, 0, stream>>>(
        h1, W2t + (size_t)i * H_ * H_, b2 + (size_t)i * H_,
        nullptr, h2, R_, H_, H_);
    // layer3: h3 = h2 @ W3 + b3  ([12928x512]x[512x128], f32 out)
    gemm_bf16_wmma<0, 0><<<dim3(CTX_ / 128, R_ / 64), 256, 0, stream>>>(
        h2, W3t + (size_t)i * CTX_ * H_, b3 + (size_t)i * CTX_,
        h3, nullptr, R_, CTX_, H_);
    // q-projection + context advance
    epilogue<<<R_ / 8, 256, 0, stream>>>(h3, ctxbuf[prev], ctxbuf[cur],
                                         Abuf, Wp, bp, qbuf, i);
  }

  // ---- final logsumexp (incl. 100 implicit zero columns) ----
  lse_kernel<<<1, 128, 0, stream>>>(qbuf, out);
}
